// QBFNNet_57784490000510
// MI455X (gfx1250) — matrix operand order repaired and checked by reading, not compiled, not run on hardware
//
#include <hip/hip_runtime.h>
#include <hip/hip_bf16.h>
#include <math.h>

#define NN 100000
#define NE 1600000
#define DD 64
#define TSTEPS 4

typedef __attribute__((ext_vector_type(16))) __bf16 v16bf;
typedef __attribute__((ext_vector_type(8)))  float  v8f;
typedef __attribute__((ext_vector_type(4)))  unsigned int u32x4;
typedef __attribute__((ext_vector_type(8)))  int i32x8;
typedef __attribute__((ext_vector_type(4)))  int i32x4;

union ABfrag { v16bf v; uint4 q[2]; };

// ---- fragment loaders (gfx1250 WMMA 16x16x32 bf16 layouts) ----

// A (16x32, MxK): lane L holds row M=L%16; hi=L/16 selects K halves.
__device__ inline v16bf load_a_bf16(const __bf16* row, int koff, int hi) {
    ABfrag f;
    f.q[0] = *(const uint4*)(row + koff + 8 * hi);
    f.q[1] = *(const uint4*)(row + koff + 16 + 8 * hi);
    return f.v;
}

__device__ inline v16bf load_a_f32(const float* row, int koff, int hi) {
    ABfrag f;
#pragma unroll
    for (int j = 0; j < 8; ++j) f.v[j]     = (__bf16)row[koff + 8 * hi + j];
#pragma unroll
    for (int j = 0; j < 8; ++j) f.v[8 + j] = (__bf16)row[koff + 16 + 8 * hi + j];
    return f.v;
}

// B (32x16, KxN): lane L holds column N; elements 0..7 = K = koff+8*hi.., 8..15 = K = koff+16+8*hi..
// Weight matrices are row-major [out=col][K] stride ld, so a B column is contiguous.
__device__ inline v16bf load_b(const __bf16* w, int ld, int col, int koff, int hi) {
    const uint4* p = (const uint4*)(w + (size_t)col * ld + koff + 8 * hi);
    ABfrag f;
    f.q[0] = p[0];
    f.q[1] = p[2];
    return f.v;
}

__device__ inline v8f wmma_bf16(v16bf a, v16bf b, v8f c) {
    return __builtin_amdgcn_wmma_f32_16x16x32_bf16(false, a, false, b, (short)0, c,
                                                   false, false);
}

__device__ inline float sigmoidf_(float x) { return 1.0f / (1.0f + __expf(-x)); }

// ---- Tensor Data Mover: stage nelem bf16 (contiguous) from global into LDS ----
// D# per CDNA5 ISA §8: group0 {count=1, lds_addr, global_addr[56:0], type=2},
// group1 {data_size=2B, tensor_dim0=nelem, tensor_dim1=1, tile_dim0=nelem,
//         tile_dim1=1, stride0=nelem}; groups 2/3 zero (<=2D tensor).
__device__ inline void tdm_stage(const __bf16* gsrc, void* lds_dst, unsigned nelem) {
    unsigned long long ga = (unsigned long long)(size_t)gsrc;
    unsigned ldsoff = (unsigned)((unsigned long long)(size_t)lds_dst & 0xffffffffull);
    u32x4 g0;
    g0[0] = 1u;                                                  // count=1 (valid)
    g0[1] = ldsoff;                                              // lds_addr (bytes)
    g0[2] = (unsigned)(ga & 0xffffffffull);                      // global_addr[31:0]
    g0[3] = (unsigned)((ga >> 32) & 0x01ffffffull) | 0x80000000u;// [56:32], type=2
    i32x8 g1;
    g1[0] = 0x00010000;                              // data_size=1 => 2 bytes
    g1[1] = (int)((nelem & 0xffffu) << 16);          // tensor_dim0[15:0]
    g1[2] = (int)(((nelem >> 16) & 0xffffu) | (1u << 16)); // dim0[31:16] | dim1=1
    g1[3] = (int)((nelem & 0xffffu) << 16);          // tile_dim0 (fits 16b)
    g1[4] = 1;                                       // tile_dim1=1, tile_dim2=0
    g1[5] = (int)nelem;                              // tensor_dim0_stride[31:0]
    g1[6] = 0;
    g1[7] = 0;
    i32x4 z4 = {0, 0, 0, 0};
#if defined(__clang_major__) && __clang_major__ >= 23
    i32x8 z8 = {0, 0, 0, 0, 0, 0, 0, 0};
    __builtin_amdgcn_tensor_load_to_lds(g0, g1, z4, z4, z8, 0);
#else
    __builtin_amdgcn_tensor_load_to_lds(g0, g1, z4, z4, 0);
#endif
}

// ---- weight conversion f32 -> bf16, packed into one ws region ----
// layout (elements): W_msg[16384] wi[12288] wh[12288] fpw1[8192] fvw1[8192]
//                    gpw1[4096] gvw1[4096]   total 65536
__global__ void k_convw(const float* wmsg, const float* wi, const float* wh,
                        const float* fpw1, const float* fvw1, const float* gpw1,
                        const float* gvw1, __bf16* dst) {
    int i = blockIdx.x * blockDim.x + threadIdx.x;
    float v;
    if      (i < 16384) v = wmsg[i];
    else if (i < 28672) v = wi[i - 16384];
    else if (i < 40960) v = wh[i - 28672];
    else if (i < 49152) v = fpw1[i - 40960];
    else if (i < 57344) v = fvw1[i - 49152];
    else if (i < 61440) v = gpw1[i - 57344];
    else if (i < 65536) v = gvw1[i - 61440];
    else return;
    dst[i] = (__bf16)v;
}

// ---- H0 init: H0[n][k] = (k%3 == node_types[n]) ----
__global__ void k_init(const int* __restrict__ node_types, __bf16* H0b, __bf16* Hb) {
    int i = blockIdx.x * blockDim.x + threadIdx.x;
    if (i >= NN * DD) return;
    int n = i >> 6, k = i & 63;
    __bf16 v = (__bf16)((k % 3 == node_types[n]) ? 1.0f : 0.0f);
    H0b[i] = v;
    Hb[i]  = v;
}

// ---- CSR-by-dst build ----
__global__ void k_hist(const int* __restrict__ dst, int* counts) {
    int e = blockIdx.x * blockDim.x + threadIdx.x;
    if (e < NE) atomicAdd(&counts[dst[e]], 1);
}

__global__ void k_blocksum(const int* __restrict__ counts, int* partial) {
    __shared__ int sm[256];
    int i = blockIdx.x * 256 + threadIdx.x;
    sm[threadIdx.x] = (i < NN) ? counts[i] : 0;
    __syncthreads();
    for (int s = 128; s > 0; s >>= 1) {
        if (threadIdx.x < s) sm[threadIdx.x] += sm[threadIdx.x + s];
        __syncthreads();
    }
    if (threadIdx.x == 0) partial[blockIdx.x] = sm[0];
}

__global__ void k_scanpartial(int* partial, int nb) {
    __shared__ int sm[512];
    int t = threadIdx.x;
    sm[t] = (t < nb) ? partial[t] : 0;
    __syncthreads();
    for (int off = 1; off < 512; off <<= 1) {
        int v = (t >= off) ? sm[t - off] : 0;
        __syncthreads();
        sm[t] += v;
        __syncthreads();
    }
    if (t < nb) partial[t] = (t == 0) ? 0 : sm[t - 1];  // exclusive
}

__global__ void k_scanfinal(const int* __restrict__ counts, const int* __restrict__ partial,
                            int* row_ptr, int* cursor) {
    __shared__ int sm[256];
    int i = blockIdx.x * 256 + threadIdx.x;
    int c = (i < NN) ? counts[i] : 0;
    sm[threadIdx.x] = c;
    __syncthreads();
    for (int off = 1; off < 256; off <<= 1) {
        int v = (threadIdx.x >= off) ? sm[threadIdx.x - off] : 0;
        __syncthreads();
        sm[threadIdx.x] += v;
        __syncthreads();
    }
    int excl = sm[threadIdx.x] - c;
    if (i < NN) {
        int rp = partial[blockIdx.x] + excl;
        row_ptr[i] = rp;
        cursor[i]  = rp;
        if (i == NN - 1) row_ptr[NN] = rp + c;
    }
}

__global__ void k_scatter(const int* __restrict__ src, const int* __restrict__ dst,
                          const int* __restrict__ et, int* cursor, int* ein) {
    int e = blockIdx.x * blockDim.x + threadIdx.x;
    if (e >= NE) return;
    int pos  = atomicAdd(&cursor[dst[e]], 1);
    ein[pos] = (src[e] << 2) | et[e];  // pack (src, etype); order in bucket irrelevant
}

// ---- typed message GEMM: X[t] = Hb @ Wmsg[t]^T + b, stored bf16 [4][N][64] ----
// Persistent: 128 threads = 4 waves (wave -> type t); Wmsg staged to LDS by TDM.
// launch_bounds(...,1): allow full VGPR budget so hoisted B fragments don't spill.
__global__ __launch_bounds__(128, 1) void k_gemm_msg(
        const __bf16* __restrict__ Hb, const __bf16* __restrict__ Wg,
        const float* __restrict__ bmsg, __bf16* __restrict__ Xb) {
    __shared__ __align__(16) __bf16 sW[16384];           // 4*64*64 bf16 = 32KB
    if (threadIdx.x < 32) tdm_stage(Wg, (void*)sW, 16384);
    __builtin_amdgcn_s_wait_tensorcnt(0);
    __syncthreads();

    int t    = threadIdx.x >> 5;
    int lane = threadIdx.x & 31;
    int hi = lane >> 4, ln = lane & 15;
    const __bf16* W = sW + (size_t)t * DD * DD;

    // loop-invariant: keep all 8 B fragments in registers across slabs
    v16bf b0[4], b1[4];
    float bias[4];
#pragma unroll
    for (int c = 0; c < 4; ++c) {
        b0[c] = load_b(W, DD, c * 16 + ln, 0, hi);
        b1[c] = load_b(W, DD, c * 16 + ln, 32, hi);
        bias[c] = bmsg[t * DD + c * 16 + ln];
    }

    for (int slab = blockIdx.x; slab < NN / 16; slab += gridDim.x) {
        int m0 = slab * 16;
        const __bf16* arow = Hb + (size_t)(m0 + ln) * DD;
        v16bf a0 = load_a_bf16(arow, 0, hi);
        v16bf a1 = load_a_bf16(arow, 32, hi);
#pragma unroll
        for (int c = 0; c < 4; ++c) {
            v8f acc = {};
            acc = wmma_bf16(a0, b0[c], acc);
            acc = wmma_bf16(a1, b1[c], acc);
#pragma unroll
            for (int v = 0; v < 8; ++v) {
                int row = m0 + v + 8 * hi;
                Xb[((size_t)t * NN + row) * DD + c * 16 + ln] = (__bf16)(acc[v] + bias[c]);
            }
        }
    }
}

// ---- atomic-free segment sum: one wave per destination node ----
__global__ void k_gather(const __bf16* __restrict__ Xb, const int* __restrict__ row_ptr,
                         const int* __restrict__ ein, float* __restrict__ mbuf) {
    int wid  = (blockIdx.x * blockDim.x + threadIdx.x) >> 5;
    int lane = threadIdx.x & 31;
    if (wid >= NN) return;
    int beg = row_ptr[wid], end = row_ptr[wid + 1];
    float a0 = 0.f, a1 = 0.f;
    for (int i = beg; i < end; ++i) {
        int v = ein[i];                       // wave-uniform
        int s = v >> 2, et = v & 3;
        const __bf16* row = Xb + ((size_t)et * NN + s) * DD;
        a0 += (float)row[lane * 2];           // 128B row, coalesced across wave
        a1 += (float)row[lane * 2 + 1];
    }
    mbuf[(size_t)wid * DD + lane * 2]     = a0;
    mbuf[(size_t)wid * DD + lane * 2 + 1] = a1;
}

// ---- fused GRU: wave computes 16 nodes x 16 features; Wi|Wh staged via TDM ----
__global__ __launch_bounds__(256, 1) void k_gru(
        const __bf16* __restrict__ Hb, const float* __restrict__ mbuf,
        const __bf16* __restrict__ WiWhG,   // Wi followed by Wh (contig)
        const float* __restrict__ bi, const float* __restrict__ bh,
        __bf16* __restrict__ HbN) {
    __shared__ __align__(16) __bf16 sW[24576];           // 48KB: Wi[12288] Wh[12288]
    if (threadIdx.x < 32) tdm_stage(WiWhG, (void*)sW, 24576);
    __builtin_amdgcn_s_wait_tensorcnt(0);
    __syncthreads();
    const __bf16* sWi = sW;
    const __bf16* sWh = sW + 12288;

    int lane = threadIdx.x & 31;
    int hi = lane >> 4, ln = lane & 15;
    int wavesPerGrid = gridDim.x * (blockDim.x >> 5);
    int gw0 = blockIdx.x * (blockDim.x >> 5) + (threadIdx.x >> 5);

    for (int gw = gw0; gw < (NN / 16) * 4; gw += wavesPerGrid) {
        int slab = gw >> 2, c = gw & 3;
        int m0 = slab * 16;
        const __bf16* hrow = Hb + (size_t)(m0 + ln) * DD;
        const float*  mrow = mbuf + (size_t)(m0 + ln) * DD;
        v16bf ah0 = load_a_bf16(hrow, 0, hi), ah1 = load_a_bf16(hrow, 32, hi);
        v16bf am0 = load_a_f32(mrow, 0, hi),  am1 = load_a_f32(mrow, 32, hi);

        v8f giS[3], ghS[3];
#pragma unroll
        for (int seg = 0; seg < 3; ++seg) {
            int col = seg * 64 + c * 16 + ln;   // 0..191 in [r|z|n] stacking
            v8f gi = {};
            gi = wmma_bf16(ah0, load_b(sWi, DD, col, 0, hi), gi);
            gi = wmma_bf16(ah1, load_b(sWi, DD, col, 32, hi), gi);
            v8f gh = {};
            gh = wmma_bf16(am0, load_b(sWh, DD, col, 0, hi), gh);
            gh = wmma_bf16(am1, load_b(sWh, DD, col, 32, hi), gh);
            float bbi = bi[col], bbh = bh[col];
#pragma unroll
            for (int v = 0; v < 8; ++v) { gi[v] += bbi; gh[v] += bbh; }
            giS[seg] = gi;
            ghS[seg] = gh;
        }
#pragma unroll
        for (int v = 0; v < 8; ++v) {
            int row = m0 + v + 8 * hi;
            int f   = c * 16 + ln;
            float r  = sigmoidf_(giS[0][v] + ghS[0][v]);
            float z  = sigmoidf_(giS[1][v] + ghS[1][v]);
            float n  = tanhf(giS[2][v] + r * ghS[2][v]);
            float mv = mbuf[(size_t)row * DD + f];
            HbN[(size_t)row * DD + f] = (__bf16)((1.0f - z) * n + z * mv);
        }
    }
}

// ---- fused heads: wave per 16-node slab; all 4 layer-1 weights staged via TDM ----
__global__ __launch_bounds__(256, 1) void k_heads(
        const __bf16* __restrict__ H0b, const __bf16* __restrict__ Hb,
        const __bf16* __restrict__ HW,   // fpw1|fvw1|gpw1|gvw1 contig
        const float* fpb1, const float* fpw2, const float* fpb2,
        const float* gpb1, const float* gpw2, const float* gpb2,
        const float* fvb1, const float* fvw2, const float* fvb2,
        const float* gvb1, const float* gvw2, const float* gvb2,
        float* accum) {
    __shared__ __align__(16) __bf16 sW[24576];           // 48KB
    if (threadIdx.x < 32) tdm_stage(HW, (void*)sW, 24576);
    __builtin_amdgcn_s_wait_tensorcnt(0);
    __syncthreads();
    const __bf16* FPW1 = sW;                  // [64][128]
    const __bf16* FVW1 = sW + 8192;           // [64][128]
    const __bf16* GPW1 = sW + 16384;          // [64][64]
    const __bf16* GVW1 = sW + 20480;          // [64][64]

    int lane = threadIdx.x & 31;
    int hi = lane >> 4, ln = lane & 15;
    int wavesPerGrid = gridDim.x * (blockDim.x >> 5);
    int gw0 = blockIdx.x * (blockDim.x >> 5) + (threadIdx.x >> 5);

    for (int gw = gw0; gw < NN / 16; gw += wavesPerGrid) {
        int m0 = gw * 16;
        const __bf16* h0row = H0b + (size_t)(m0 + ln) * DD;
        const __bf16* hrow  = Hb + (size_t)(m0 + ln) * DD;
        v16bf a00 = load_a_bf16(h0row, 0, hi), a01 = load_a_bf16(h0row, 32, hi);
        v16bf ah0 = load_a_bf16(hrow, 0, hi),  ah1 = load_a_bf16(hrow, 32, hi);

        float pf0[8], pf1[8], pg0[8], pg1[8], vf[8], vg[8];
#pragma unroll
        for (int v = 0; v < 8; ++v) { pf0[v] = pf1[v] = pg0[v] = pg1[v] = vf[v] = vg[v] = 0.f; }

#pragma unroll
        for (int c = 0; c < 4; ++c) {
            int col = c * 16 + ln;
            // fp layer1: K=128 over [H0 | H]
            v8f af = {};
            af = wmma_bf16(a00, load_b(FPW1, 128, col, 0, hi), af);
            af = wmma_bf16(a01, load_b(FPW1, 128, col, 32, hi), af);
            af = wmma_bf16(ah0, load_b(FPW1, 128, col, 64, hi), af);
            af = wmma_bf16(ah1, load_b(FPW1, 128, col, 96, hi), af);
            float fb = fpb1[col], fw0 = fpw2[col], fw1 = fpw2[64 + col];
#pragma unroll
            for (int v = 0; v < 8; ++v) {
                float t = tanhf(af[v] + fb);
                pf0[v] += t * fw0; pf1[v] += t * fw1;
            }
            // gp layer1: K=64 over H
            v8f ag = {};
            ag = wmma_bf16(ah0, load_b(GPW1, DD, col, 0, hi), ag);
            ag = wmma_bf16(ah1, load_b(GPW1, DD, col, 32, hi), ag);
            float gb = gpb1[col], gw0v = gpw2[col], gw1v = gpw2[64 + col];
#pragma unroll
            for (int v = 0; v < 8; ++v) {
                float t = tanhf(ag[v] + gb);
                pg0[v] += t * gw0v; pg1[v] += t * gw1v;
            }
            // fv layer1: K=128
            v8f av = {};
            av = wmma_bf16(a00, load_b(FVW1, 128, col, 0, hi), av);
            av = wmma_bf16(a01, load_b(FVW1, 128, col, 32, hi), av);
            av = wmma_bf16(ah0, load_b(FVW1, 128, col, 64, hi), av);
            av = wmma_bf16(ah1, load_b(FVW1, 128, col, 96, hi), av);
            float vb = fvb1[col], vw0 = fvw2[col];
#pragma unroll
            for (int v = 0; v < 8; ++v) vf[v] += tanhf(av[v] + vb) * vw0;
            // gv layer1: K=64
            v8f au = {};
            au = wmma_bf16(ah0, load_b(GVW1, DD, col, 0, hi), au);
            au = wmma_bf16(ah1, load_b(GVW1, DD, col, 32, hi), au);
            float ub = gvb1[col], uw0 = gvw2[col];
#pragma unroll
            for (int v = 0; v < 8; ++v) vg[v] += tanhf(au[v] + ub) * uw0;
        }

        float s0 = 0.f, s1 = 0.f, sv = 0.f;
#pragma unroll
        for (int v = 0; v < 8; ++v) {
            float f0 = pf0[v], f1 = pf1[v], g0 = pg0[v], g1 = pg1[v], fvv = vf[v], gvv = vg[v];
#pragma unroll
            for (int off = 8; off >= 1; off >>= 1) {   // reduce over 16 columns
                f0  += __shfl_xor(f0, off, 16);
                f1  += __shfl_xor(f1, off, 16);
                g0  += __shfl_xor(g0, off, 16);
                g1  += __shfl_xor(g1, off, 16);
                fvv += __shfl_xor(fvv, off, 16);
                gvv += __shfl_xor(gvv, off, 16);
            }
            f0 += fpb2[0]; f1 += fpb2[1];
            g0 += gpb2[0]; g1 += gpb2[1];
            fvv += fvb2[0]; gvv += gvb2[0];
            if (ln == 0) {                              // one lane per node-group
                s0 += sigmoidf_(f0) * g0;
                s1 += sigmoidf_(f1) * g1;
                sv += sigmoidf_(fvv) * gvv;
            }
        }
        s0 += __shfl_xor(s0, 16, 32);
        s1 += __shfl_xor(s1, 16, 32);
        sv += __shfl_xor(sv, 16, 32);
        if (lane == 0) {
            atomicAdd(&accum[0], s0);
            atomicAdd(&accum[1], s1);
            atomicAdd(&accum[2], sv);
        }
    }
}

__global__ void k_final(const float* __restrict__ accum, float* __restrict__ out) {
    if (threadIdx.x == 0 && blockIdx.x == 0) {
        float p0 = accum[0], p1 = accum[1], vv = accum[2];
        float mx  = fmaxf(p0, p1);
        float lse = mx + logf(__expf(p0 - mx) + __expf(p1 - mx));
        out[0] = p0 - lse;
        out[1] = p1 - lse;
        out[2] = tanhf(vv);
    }
}

extern "C" void kernel_launch(void* const* d_in, const int* in_sizes, int n_in,
                              void* d_out, int out_size, void* d_ws, size_t ws_size,
                              hipStream_t stream) {
    const int*   node_types = (const int*)d_in[0];
    const int*   src    = (const int*)d_in[1];
    const int*   dst    = (const int*)d_in[2];
    const int*   etype  = (const int*)d_in[3];
    const float* W_msg  = (const float*)d_in[4];
    const float* b_msg  = (const float*)d_in[5];
    const float* gru_wi = (const float*)d_in[6];
    const float* gru_wh = (const float*)d_in[7];
    const float* gru_bi = (const float*)d_in[8];
    const float* gru_bh = (const float*)d_in[9];
    const float* fp_w1 = (const float*)d_in[10];
    const float* fp_b1 = (const float*)d_in[11];
    const float* fp_w2 = (const float*)d_in[12];
    const float* fp_b2 = (const float*)d_in[13];
    const float* gp_w1 = (const float*)d_in[14];
    const float* gp_b1 = (const float*)d_in[15];
    const float* gp_w2 = (const float*)d_in[16];
    const float* gp_b2 = (const float*)d_in[17];
    const float* fv_w1 = (const float*)d_in[18];
    const float* fv_b1 = (const float*)d_in[19];
    const float* fv_w2 = (const float*)d_in[20];
    const float* fv_b2 = (const float*)d_in[21];
    const float* gv_w1 = (const float*)d_in[22];
    const float* gv_b1 = (const float*)d_in[23];
    const float* gv_w2 = (const float*)d_in[24];
    const float* gv_b2 = (const float*)d_in[25];

    char*  ws  = (char*)d_ws;
    size_t off = 0;
    auto alloc = [&](size_t bytes) -> void* {
        void* p = ws + off;
        off = (off + bytes + 255) & ~(size_t)255;
        return p;
    };

    __bf16* H0b    = (__bf16*)alloc((size_t)NN * DD * 2);
    __bf16* HbA    = (__bf16*)alloc((size_t)NN * DD * 2);
    __bf16* HbB    = (__bf16*)alloc((size_t)NN * DD * 2);
    __bf16* Xb     = (__bf16*)alloc((size_t)4 * NN * DD * 2);
    float*  mbuf   = (float*)alloc((size_t)NN * DD * 4);
    __bf16* Wb     = (__bf16*)alloc((size_t)65536 * 2);
    int*    counts = (int*)alloc((size_t)NN * 4);
    int*    cursor = (int*)alloc((size_t)NN * 4);
    int*    rowptr = (int*)alloc((size_t)(NN + 1) * 4);
    int*    partial= (int*)alloc((size_t)512 * 4);
    int*    ein    = (int*)alloc((size_t)NE * 4);
    float*  accum  = (float*)alloc((size_t)16);

    const __bf16* WmsgB = Wb;            // [16384]
    const __bf16* WiWhB = Wb + 16384;    // Wi[12288] Wh[12288] contiguous
    const __bf16* HeadW = Wb + 40960;    // fpw1|fvw1|gpw1|gvw1 = 24576 contiguous

    const int NB = (NN + 255) / 256;     // 391

    hipMemsetAsync(counts, 0, (size_t)NN * 4, stream);
    hipMemsetAsync(accum, 0, 16, stream);

    k_convw<<<256, 256, 0, stream>>>(W_msg, gru_wi, gru_wh, fp_w1, fv_w1, gp_w1, gv_w1, Wb);
    k_init<<<(NN * DD + 255) / 256, 256, 0, stream>>>(node_types, H0b, HbA);

    // build dst-CSR (once per call; reused by all 4 steps)
    k_hist<<<(NE + 255) / 256, 256, 0, stream>>>(dst, counts);
    k_blocksum<<<NB, 256, 0, stream>>>(counts, partial);
    k_scanpartial<<<1, 512, 0, stream>>>(partial, NB);
    k_scanfinal<<<NB, 256, 0, stream>>>(counts, partial, rowptr, cursor);
    k_scatter<<<(NE + 255) / 256, 256, 0, stream>>>(src, dst, etype, cursor, ein);

    const __bf16* Hcur = HbA;
    __bf16*       Hnxt = HbB;
    for (int t = 0; t < TSTEPS; ++t) {
        k_gemm_msg<<<640, 128, 0, stream>>>(Hcur, WmsgB, b_msg, Xb);
        k_gather<<<(NN * 32 + 255) / 256, 256, 0, stream>>>(Xb, rowptr, ein, mbuf);
        k_gru<<<768, 256, 0, stream>>>(Hcur, mbuf, WiWhB, gru_bi, gru_bh, Hnxt);
        const __bf16* tmp = Hcur;
        Hcur = Hnxt;
        Hnxt = (__bf16*)tmp;
    }

    k_heads<<<384, 256, 0, stream>>>(
        H0b, Hcur, HeadW,
        fp_b1, fp_w2, fp_b2,
        gp_b1, gp_w2, gp_b2,
        fv_b1, fv_w2, fv_b2,
        gv_b1, gv_w2, gv_b2,
        accum);
    k_final<<<1, 64, 0, stream>>>(accum, (float*)d_out);
}
